// MemoryTransformer_8881992368294
// MI455X (gfx1250) — matrix operand order repaired and checked by reading, not compiled, not run on hardware
//
#include <hip/hip_runtime.h>
#include <hip/hip_bf16.h>

// ---------------------------------------------------------------------------
// Transformer forward for MI455X (gfx1250): f16 WMMA + TDM double-buffered
// tile staging. B=4 S=2048 D=768 H=12 HD=64 L=6 V=32000 FF=3072
// ---------------------------------------------------------------------------

constexpr int Bc = 4, Sc = 2048, Dc = 768, Hc = 12, Lc = 6, Vc = 32000, FFc = 3072;

typedef __attribute__((ext_vector_type(16))) _Float16 v16h;
typedef __attribute__((ext_vector_type(8)))  float    v8f;
typedef __attribute__((ext_vector_type(4)))  unsigned int u32x4;
typedef __attribute__((ext_vector_type(8)))  int          i32x8;
typedef __attribute__((ext_vector_type(4)))  int          i32x4;

__device__ __forceinline__ v8f wmma_f16(v16h a, v16h b, v8f c) {
  return __builtin_amdgcn_wmma_f32_16x16x32_f16(false, a, false, b, (short)0, c,
                                                false, false);
}

// ---------------------------------------------------------------------------
// Tensor Data Mover: 2D tile load Global -> LDS (D# per ISA 08 sec.8).
// data_size=2B. LDS padding (interval/amount codes) builds the bank-conflict-
// free padded row strides directly.
// ---------------------------------------------------------------------------
#if defined(__gfx1250__) && __has_builtin(__builtin_amdgcn_tensor_load_to_lds) && \
    __has_builtin(__builtin_amdgcn_s_wait_tensorcnt)
#define HAVE_TDM 1
#endif

__device__ __forceinline__ void tdm_load_2d_f16(unsigned lds_off, const void* gaddr,
                                                int tile_d0, int tile_d1,
                                                unsigned tensor_d0, unsigned tensor_d1,
                                                unsigned stride_elts,
                                                int pad_interval_code,
                                                int pad_amount_code) {
#ifdef HAVE_TDM
  unsigned long long ga = (unsigned long long)gaddr;
  u32x4 g0 = {
      1u,                                   // count=1 valid user descriptor
      lds_off,                              // lds_addr (bytes)
      (unsigned)(ga & 0xFFFFFFFFu),         // global_addr[31:0]
      (((unsigned)(ga >> 32)) & 0x01FFFFFFu) | (2u << 30)  // addr[56:32] | type=2
  };
  unsigned w0 = (1u << 16);                 // data_size = 2 bytes
  if (pad_amount_code >= 0)
    w0 |= (1u << 20) | ((unsigned)pad_interval_code << 22) |
          ((unsigned)pad_amount_code << 25);
  i32x8 g1 = {
      (int)w0,
      (int)((tensor_d0 & 0xFFFFu) << 16),                       // dim0[15:0]
      (int)((tensor_d0 >> 16) | ((tensor_d1 & 0xFFFFu) << 16)), // dim0 hi | dim1 lo
      (int)((tensor_d1 >> 16) | ((unsigned)tile_d0 << 16)),     // dim1 hi | tile0
      tile_d1,                                                  // tile1 (tile2=0)
      (int)stride_elts,                                         // dim0_stride[31:0]
      0, 0
  };
  i32x4 z4 = {0, 0, 0, 0};
#if __clang_major__ >= 23
  i32x8 z8 = {0, 0, 0, 0, 0, 0, 0, 0};
  __builtin_amdgcn_tensor_load_to_lds(g0, g1, z4, z4, z8, 0);
#else
  __builtin_amdgcn_tensor_load_to_lds(g0, g1, z4, z4, 0);
#endif
#else
  (void)lds_off; (void)gaddr; (void)tile_d0; (void)tile_d1;
  (void)tensor_d0; (void)tensor_d1; (void)stride_elts;
  (void)pad_interval_code; (void)pad_amount_code;
#endif
}

// ---------------------------------------------------------------------------
// DPP row_ror butterfly reductions over 16-lane rows (all lanes get result).
// ---------------------------------------------------------------------------
#define DPP_ROR16(x, n)                                                        \
  __int_as_float(__builtin_amdgcn_update_dpp(0, __float_as_int(x),             \
                                             0x120 + (n), 0xF, 0xF, true))
__device__ __forceinline__ float red16_max(float x) {
  x = fmaxf(x, DPP_ROR16(x, 1));
  x = fmaxf(x, DPP_ROR16(x, 2));
  x = fmaxf(x, DPP_ROR16(x, 4));
  x = fmaxf(x, DPP_ROR16(x, 8));
  return x;
}
__device__ __forceinline__ float red16_sum(float x) {
  x += DPP_ROR16(x, 1);
  x += DPP_ROR16(x, 2);
  x += DPP_ROR16(x, 4);
  x += DPP_ROR16(x, 8);
  return x;
}

// ---------------------------------------------------------------------------
// fp32 -> f16 elementwise (embedding table)
// ---------------------------------------------------------------------------
__global__ __launch_bounds__(256) void k_f32_to_f16(const float* __restrict__ src,
                                                    _Float16* __restrict__ dst,
                                                    size_t n) {
  size_t i = (size_t)blockIdx.x * 256 + threadIdx.x;
  if (i < n) dst[i] = (_Float16)src[i];
}

// ---------------------------------------------------------------------------
// fp32 [Kd,Nd] -> f16 [Nd,Kd] transpose (per-layer, LDS tiled, coalesced)
// ---------------------------------------------------------------------------
__global__ __launch_bounds__(256) void k_transpose_f16(const float* __restrict__ src,
                                                       _Float16* __restrict__ dst,
                                                       int Kd, int Nd) {
  __shared__ float tile[32][33];
  const float* W = src + (size_t)blockIdx.z * Kd * Nd;
  _Float16* WT = dst + (size_t)blockIdx.z * Kd * Nd;
  const int n0 = blockIdx.x * 32, k0 = blockIdx.y * 32;
  const int tx = threadIdx.x & 31, ty = threadIdx.x >> 5;
#pragma unroll
  for (int j = 0; j < 4; ++j) {
    int kk = ty + j * 8;
    tile[kk][tx] = W[(size_t)(k0 + kk) * Nd + n0 + tx];
  }
  __syncthreads();
#pragma unroll
  for (int j = 0; j < 4; ++j) {
    int nn = ty + j * 8;
    WT[(size_t)(n0 + nn) * Kd + k0 + tx] = (_Float16)tile[tx][nn];
  }
}

// ---------------------------------------------------------------------------
// Embedding gather + RoPE (interleaved pairs), fp32 out. One block per token.
// ---------------------------------------------------------------------------
__global__ __launch_bounds__(256) void k_embed_rope(const int* __restrict__ ids,
                                                    const float* __restrict__ emb,
                                                    float* __restrict__ h) {
  const int bs = blockIdx.x;
  const int s  = bs % Sc;
  const int tok = ids[bs];
  const float* e = emb + (size_t)tok * Dc;
  float* hr = h + (size_t)bs * Dc;
  for (int p = threadIdx.x; p < Dc / 2; p += 256) {
    float inv = __expf(-(float)(2 * p) / (float)Dc * 9.210340372f);
    float ang = (float)s * inv;
    float sn, cs;
    __sincosf(ang, &sn, &cs);
    float xe = e[2 * p], xo = e[2 * p + 1];
    hr[2 * p]     = xe * cs - xo * sn;
    hr[2 * p + 1] = xo * cs + xe * sn;
  }
}

// ---------------------------------------------------------------------------
// LayerNorm: fp32 in, f16 out. One block (256 thr) per row of 768.
// ---------------------------------------------------------------------------
__global__ __launch_bounds__(256) void k_layernorm(const float* __restrict__ x,
                                                   const float* __restrict__ w,
                                                   const float* __restrict__ b,
                                                   _Float16* __restrict__ out) {
  __shared__ float s1[256], s2[256];
  const int row = blockIdx.x;
  const float* xr = x + (size_t)row * Dc;
  float vals[3];
  float a0 = 0.f, a1 = 0.f;
#pragma unroll
  for (int j = 0; j < 3; ++j) {
    vals[j] = xr[threadIdx.x + j * 256];
    a0 += vals[j];
    a1 += vals[j] * vals[j];
  }
  s1[threadIdx.x] = a0;
  s2[threadIdx.x] = a1;
  __syncthreads();
  for (int off = 128; off > 0; off >>= 1) {
    if ((int)threadIdx.x < off) {
      s1[threadIdx.x] += s1[threadIdx.x + off];
      s2[threadIdx.x] += s2[threadIdx.x + off];
    }
    __syncthreads();
  }
  const float mean = s1[0] * (1.f / Dc);
  const float var  = s2[0] * (1.f / Dc) - mean * mean;
  const float rstd = rsqrtf(var + 1e-5f);
#pragma unroll
  for (int j = 0; j < 3; ++j) {
    int d = threadIdx.x + j * 256;
    out[(size_t)row * Dc + d] = (_Float16)((vals[j] - mean) * rstd * w[d] + b[d]);
  }
}

// ---------------------------------------------------------------------------
// WMMA GEMM: C[M,N] = epilogue(A[M,K](f16) @ WT^T + bias), WT is [N,K] f16.
// Block tile 256x64, 256 threads = 8 waves; each wave: 32 rows x 64 cols
// (2 A-frags x 4 B-frags -> 8 WMMAs per 32-wide K step).
// Double-buffered LDS: TDM (or plain uint4 copies) for tile i+1 overlap the WMMA
// compute of tile i; one barrier per K step.
// ---------------------------------------------------------------------------
template <int GELU, int RES, int NT>
__global__ __launch_bounds__(256) void k_gemm(const _Float16* __restrict__ A,
                                              const _Float16* __restrict__ WT,
                                              const float* __restrict__ bias,
                                              const float* __restrict__ resid,
                                              float* __restrict__ outF,
                                              _Float16* __restrict__ outH,
                                              int M, int N, int K) {
  __shared__ _Float16 As[2][256 * 40];   // 256 rows, stride 40 (pad 8)
  __shared__ _Float16 BsT[2][64 * 40];   // 64 n-rows, stride 40

  const int tid  = threadIdx.x;
  const int wave = tid >> 5;
  const int lane = tid & 31;
  const int hh   = lane >> 4;
  const int l16  = lane & 15;
  const int kb   = hh * 8;
  const int n0 = blockIdx.x * 64;
  const int m0 = blockIdx.y * 256;

  auto stage = [&](int k0, int buf) {
#ifdef HAVE_TDM
    if (wave == 0) {
      // A tile: 256 rows x 32 f16; pad 4 dwords per 16 dwords -> stride 40
      tdm_load_2d_f16((unsigned)(size_t)&As[buf][0], &A[(size_t)m0 * K + k0],
                      32, 256, (unsigned)K, (unsigned)M, (unsigned)K, 3, 3);
      // B tile: 64 n-rows x 32 f16 from WT[N,K]
      tdm_load_2d_f16((unsigned)(size_t)&BsT[buf][0], &WT[(size_t)n0 * K + k0],
                      32, 64, (unsigned)K, (unsigned)N, (unsigned)K, 3, 3);
    }
#else
#pragma unroll
    for (int c = 0; c < 4; ++c) {
      int chunk = tid + c * 256;
      int row = chunk >> 2;
      int col = (chunk & 3) * 8;
      *(uint4*)&As[buf][row * 40 + col] =
          *(const uint4*)&A[(size_t)(m0 + row) * K + k0 + col];
    }
    {
      int n  = tid >> 2;
      int kc = (tid & 3) * 8;
      *(uint4*)&BsT[buf][n * 40 + kc] =
          *(const uint4*)&WT[(size_t)(n0 + n) * K + k0 + kc];
    }
    if (k0 + 32 < K)
      __builtin_prefetch(&A[(size_t)(m0 + tid) * K + k0 + 32], 0, 3);
#endif
  };

  v8f acc[8];
#pragma unroll
  for (int i = 0; i < 8; ++i) acc[i] = {};

  const int nk = K / 32;
  stage(0, 0);
  for (int it = 0; it < nk; ++it) {
#ifdef HAVE_TDM
    if (wave == 0) __builtin_amdgcn_s_wait_tensorcnt((short)0);
#endif
    __syncthreads();                       // publish tile it
    if (it + 1 < nk) stage((it + 1) * 32, (it + 1) & 1);  // overlaps compute
    const _Float16* __restrict__ Ab = &As[it & 1][0];
    const _Float16* __restrict__ Bb = &BsT[it & 1][0];

    v16h af[2];
#pragma unroll
    for (int f = 0; f < 2; ++f)
#pragma unroll
      for (int e = 0; e < 16; ++e) {
        int kk = kb + e + (e >= 8 ? 8 : 0);
        af[f][e] = Ab[(wave * 32 + f * 16 + l16) * 40 + kk];
      }
#pragma unroll
    for (int nt = 0; nt < 4; ++nt) {
      v16h bf;
#pragma unroll
      for (int e = 0; e < 16; ++e)
        bf[e] = Bb[(nt * 16 + l16) * 40 + kb + e + (e >= 8 ? 8 : 0)];
      acc[nt]     = wmma_f16(af[0], bf, acc[nt]);
      acc[4 + nt] = wmma_f16(af[1], bf, acc[4 + nt]);
    }
  }

#pragma unroll
  for (int f = 0; f < 2; ++f)
#pragma unroll
    for (int nt = 0; nt < 4; ++nt) {
      int col = n0 + nt * 16 + l16;
      float bb = bias ? bias[col] : 0.f;
#pragma unroll
      for (int e = 0; e < 8; ++e) {
        int row = m0 + wave * 32 + f * 16 + e + hh * 8;
        float x = acc[f * 4 + nt][e] + bb;
        if (GELU) x = 0.5f * x * (1.f + erff(x * 0.70710678118f));
        if (RES)  x += resid[(size_t)row * N + col];
        if (outF) {
          if (NT) __builtin_nontemporal_store(x, &outF[(size_t)row * N + col]);
          else    outF[(size_t)row * N + col] = x;
        }
        if (outH) outH[(size_t)row * N + col] = (_Float16)x;
      }
    }
}

// ---------------------------------------------------------------------------
// Flash attention: q,k,v f16 [B*S, D]. 128 threads = 4 waves, each wave one
// 16-query tile; 32 keys per step. Double-buffered K/V tiles: K via TDM
// (padded stride 72), V transposed into LDS (manual scatter) so P@V B-frags
// are contiguous b128 reads. One barrier per kv step.
// ---------------------------------------------------------------------------
__global__ __launch_bounds__(128) void k_attention(const _Float16* __restrict__ q,
                                                   const _Float16* __restrict__ k,
                                                   const _Float16* __restrict__ v,
                                                   _Float16* __restrict__ ctx) {
  __shared__ _Float16 Ks[2][32 * 72];      // [key][dim], stride 72 (pad 8)
  __shared__ _Float16 VsT[2][64 * 40];     // [dim][key], stride 40
  __shared__ _Float16 Ps[4 * 16 * 40];     // per-wave P tile, stride 40

  const int bh = blockIdx.y;
  const int b  = bh / Hc;
  const int hd = bh % Hc;
  const int q0 = blockIdx.x * 64;
  const int tid  = threadIdx.x;
  const int wave = tid >> 5;
  const int lane = tid & 31;
  const int hh   = lane >> 4;
  const int l16  = lane & 15;
  const int kb   = hh * 8;
  const int qrow0 = q0 + wave * 16;

  auto stage = [&](int kv0, int buf) {
#ifdef HAVE_TDM
    if (wave == 0) {
      // K tile 32 rows x 64 f16; pad 4 dwords per 32 dwords -> stride 72
      tdm_load_2d_f16((unsigned)(size_t)&Ks[buf][0],
                      &k[((size_t)(b * Sc + kv0)) * Dc + hd * 64],
                      64, 32, (unsigned)Dc, (unsigned)(Bc * Sc), (unsigned)Dc,
                      4, 3);
    }
#else
#pragma unroll
    for (int c = 0; c < 2; ++c) {
      int chunk = tid + c * 128;
      int row = chunk >> 3;
      int col = (chunk & 7) * 8;
      *(uint4*)&Ks[buf][row * 72 + col] =
          *(const uint4*)&k[((size_t)(b * Sc + kv0 + row)) * Dc + hd * 64 + col];
    }
#endif
    // V tile transposed into LDS (manual scatter; TDM cannot transpose)
#pragma unroll
    for (int c = 0; c < 2; ++c) {
      int chunk = tid + c * 128;
      int row = chunk >> 3;                 // key 0..31
      int col = (chunk & 7) * 8;            // dim
      union { uint4 u; _Float16 e[8]; } t;
      t.u = *(const uint4*)&v[((size_t)(b * Sc + kv0 + row)) * Dc + hd * 64 + col];
#pragma unroll
      for (int j = 0; j < 8; ++j) VsT[buf][(col + j) * 40 + row] = t.e[j];
    }
  };

  v16h qa0, qa1;
  {
    size_t base = ((size_t)(b * Sc + qrow0 + l16)) * Dc + hd * 64;
#pragma unroll
    for (int e = 0; e < 16; ++e) {
      int kk = kb + e + (e >= 8 ? 8 : 0);
      qa0[e] = q[base + kk];
      qa1[e] = q[base + kk + 32];
    }
  }

  float m_r[8], l_r[8];
  v8f acc[4];
#pragma unroll
  for (int e = 0; e < 8; ++e) { m_r[e] = -1e30f; l_r[e] = 0.f; }
#pragma unroll
  for (int nt = 0; nt < 4; ++nt) acc[nt] = {};

  const int nkv = Sc / 32;
  stage(0, 0);
  for (int it = 0; it < nkv; ++it) {
#ifdef HAVE_TDM
    if (wave == 0) __builtin_amdgcn_s_wait_tensorcnt((short)0);
#endif
    __syncthreads();                       // publish tile it
    if (it + 1 < nkv) stage((it + 1) * 32, (it + 1) & 1);  // overlaps compute
    const _Float16* __restrict__ Kb = &Ks[it & 1][0];
    const _Float16* __restrict__ Vb = &VsT[it & 1][0];

    // scores: 16 q x 32 keys, K-dim = 64 (two WMMA per 16-key tile)
    v8f sc[2];
#pragma unroll
    for (int nh = 0; nh < 2; ++nh) {
      v16h bf0, bf1;
#pragma unroll
      for (int e = 0; e < 16; ++e) {
        int kd = kb + e + (e >= 8 ? 8 : 0);
        bf0[e] = Kb[(nh * 16 + l16) * 72 + kd];
        bf1[e] = Kb[(nh * 16 + l16) * 72 + kd + 32];
      }
      v8f cc = {};
      cc = wmma_f16(qa0, bf0, cc);
      cc = wmma_f16(qa1, bf1, cc);
      sc[nh] = cc;
    }

    // online softmax; row r = e + 8*hh, cols striped over 16 lanes
#pragma unroll
    for (int e = 0; e < 8; ++e) {
      float v0 = sc[0][e] * 0.125f;
      float v1 = sc[1][e] * 0.125f;
      float mnew = fmaxf(m_r[e], red16_max(fmaxf(v0, v1)));
      float p0 = __expf(v0 - mnew);
      float p1 = __expf(v1 - mnew);
      float rs = red16_sum(p0 + p1);
      float alpha = __expf(m_r[e] - mnew);
      l_r[e] = l_r[e] * alpha + rs;
      m_r[e] = mnew;
#pragma unroll
      for (int nt = 0; nt < 4; ++nt) acc[nt][e] *= alpha;
      int row = e + hh * 8;
      Ps[wave * 640 + row * 40 + l16]      = (_Float16)p0;
      Ps[wave * 640 + row * 40 + 16 + l16] = (_Float16)p1;
    }

    // P (16x32) as A fragment from this wave's private LDS region
    v16h pa;
#pragma unroll
    for (int e = 0; e < 16; ++e) {
      int kk = kb + e + (e >= 8 ? 8 : 0);
      pa[e] = Ps[wave * 640 + l16 * 40 + kk];
    }
    // ctx += P @ V (B-frags contiguous from VsT)
#pragma unroll
    for (int nt = 0; nt < 4; ++nt) {
      v16h bv;
#pragma unroll
      for (int e = 0; e < 16; ++e)
        bv[e] = Vb[(nt * 16 + l16) * 40 + kb + e + (e >= 8 ? 8 : 0)];
      acc[nt] = wmma_f16(pa, bv, acc[nt]);
    }
  }

#pragma unroll
  for (int nt = 0; nt < 4; ++nt)
#pragma unroll
    for (int e = 0; e < 8; ++e) {
      int row = e + hh * 8;
      float val = acc[nt][e] / l_r[e];
      ctx[((size_t)(b * Sc + qrow0 + row)) * Dc + hd * 64 + nt * 16 + l16] =
          (_Float16)val;
    }
}

// ---------------------------------------------------------------------------
// Host-side orchestration
// ---------------------------------------------------------------------------
extern "C" void kernel_launch(void* const* d_in, const int* in_sizes, int n_in,
                              void* d_out, int out_size, void* d_ws, size_t ws_size,
                              hipStream_t stream) {
  (void)in_sizes; (void)n_in; (void)out_size; (void)ws_size;
  const int*   ids  = (const int*)d_in[0];
  const float* emb  = (const float*)d_in[1];
  const float* q_w  = (const float*)d_in[2];
  const float* q_b  = (const float*)d_in[3];
  const float* k_w  = (const float*)d_in[4];
  const float* k_b  = (const float*)d_in[5];
  const float* v_w  = (const float*)d_in[6];
  const float* v_b  = (const float*)d_in[7];
  const float* o_w  = (const float*)d_in[8];
  const float* o_b  = (const float*)d_in[9];
  const float* f1_w = (const float*)d_in[10];
  const float* f1_b = (const float*)d_in[11];
  const float* f2_w = (const float*)d_in[12];
  const float* f2_b = (const float*)d_in[13];
  const float* n1_w = (const float*)d_in[14];
  const float* n1_b = (const float*)d_in[15];
  const float* n2_w = (const float*)d_in[16];
  const float* n2_b = (const float*)d_in[17];
  const float* fn_w = (const float*)d_in[18];
  const float* fn_b = (const float*)d_in[19];

  const size_t BS = (size_t)Bc * Sc;

  char* ws = (char*)d_ws;
  size_t off = 0;
  auto alloc = [&](size_t bytes) -> char* {
    char* p = ws + off;
    off = (off + bytes + 255) & ~(size_t)255;
    return p;
  };

  float*    h    = (float*)   alloc(BS * Dc * 4);
  _Float16* xn   = (_Float16*)alloc(BS * Dc * 2);
  _Float16* qbuf = (_Float16*)alloc(BS * Dc * 2);
  _Float16* kbuf = (_Float16*)alloc(BS * Dc * 2);
  _Float16* vbuf = (_Float16*)alloc(BS * Dc * 2);
  _Float16* cbuf = (_Float16*)alloc(BS * Dc * 2);
  _Float16* gbuf = (_Float16*)alloc(BS * FFc * 2);
  _Float16* wq16 = (_Float16*)alloc((size_t)Lc * Dc * Dc * 2);   // [N,K] per layer
  _Float16* wk16 = (_Float16*)alloc((size_t)Lc * Dc * Dc * 2);
  _Float16* wv16 = (_Float16*)alloc((size_t)Lc * Dc * Dc * 2);
  _Float16* wo16 = (_Float16*)alloc((size_t)Lc * Dc * Dc * 2);
  _Float16* wf1  = (_Float16*)alloc((size_t)Lc * Dc * FFc * 2);  // [FF,D] per layer
  _Float16* wf2  = (_Float16*)alloc((size_t)Lc * FFc * Dc * 2);  // [D,FF] per layer
  _Float16* emb16= (_Float16*)alloc((size_t)Vc * Dc * 2);        // already [N,K]

  auto trans = [&](const float* s, _Float16* d, int Kd, int Nd) {
    dim3 g(Nd / 32, Kd / 32, Lc);
    k_transpose_f16<<<g, 256, 0, stream>>>(s, d, Kd, Nd);
  };
  trans(q_w,  wq16, Dc, Dc);
  trans(k_w,  wk16, Dc, Dc);
  trans(v_w,  wv16, Dc, Dc);
  trans(o_w,  wo16, Dc, Dc);
  trans(f1_w, wf1,  Dc, FFc);
  trans(f2_w, wf2,  FFc, Dc);
  {
    size_t n = (size_t)Vc * Dc;
    k_f32_to_f16<<<dim3((unsigned)((n + 255) / 256)), 256, 0, stream>>>(emb, emb16, n);
  }

  k_embed_rope<<<dim3((unsigned)BS), 256, 0, stream>>>(ids, emb, h);

  const dim3 gD(Dc / 64, (unsigned)(BS / 256));
  const dim3 gF(FFc / 64, (unsigned)(BS / 256));
  const dim3 gA(Sc / 64, Bc * Hc);

  for (int l = 0; l < Lc; ++l) {
    k_layernorm<<<dim3((unsigned)BS), 256, 0, stream>>>(h, n1_w + l * Dc,
                                                        n1_b + l * Dc, xn);
    const _Float16* wq = wq16 + (size_t)l * Dc * Dc;
    const _Float16* wk = wk16 + (size_t)l * Dc * Dc;
    const _Float16* wv = wv16 + (size_t)l * Dc * Dc;
    const _Float16* wo = wo16 + (size_t)l * Dc * Dc;
    k_gemm<0, 0, 0><<<gD, 256, 0, stream>>>(xn, wq, q_b + l * Dc, nullptr,
                                            nullptr, qbuf, (int)BS, Dc, Dc);
    k_gemm<0, 0, 0><<<gD, 256, 0, stream>>>(xn, wk, k_b + l * Dc, nullptr,
                                            nullptr, kbuf, (int)BS, Dc, Dc);
    k_gemm<0, 0, 0><<<gD, 256, 0, stream>>>(xn, wv, v_b + l * Dc, nullptr,
                                            nullptr, vbuf, (int)BS, Dc, Dc);
    k_attention<<<gA, 128, 0, stream>>>(qbuf, kbuf, vbuf, cbuf);
    // h = h + ctx @ o_w + o_b
    k_gemm<0, 1, 0><<<gD, 256, 0, stream>>>(cbuf, wo, o_b + l * Dc, h,
                                            h, nullptr, (int)BS, Dc, Dc);
    k_layernorm<<<dim3((unsigned)BS), 256, 0, stream>>>(h, n2_w + l * Dc,
                                                        n2_b + l * Dc, xn);
    // g = gelu(xn @ f1_w + f1_b)
    k_gemm<1, 0, 0><<<gF, 256, 0, stream>>>(xn, wf1 + (size_t)l * Dc * FFc,
                                            f1_b + l * FFc, nullptr,
                                            nullptr, gbuf, (int)BS, FFc, Dc);
    // h = h + g @ f2_w + f2_b
    k_gemm<0, 1, 0><<<gD, 256, 0, stream>>>(gbuf, wf2 + (size_t)l * FFc * Dc,
                                            f2_b + l * Dc, h,
                                            h, nullptr, (int)BS, Dc, FFc);
  }

  k_layernorm<<<dim3((unsigned)BS), 256, 0, stream>>>(h, fn_w, fn_b, xn);
  // logits = xn @ emb^T (emb16 is [V,D] = [N,K] natively); non-temporal stores
  const dim3 gV(Vc / 64, (unsigned)(BS / 256));
  k_gemm<0, 0, 1><<<gV, 256, 0, stream>>>(xn, emb16, nullptr, nullptr,
                                          (float*)d_out, nullptr,
                                          (int)BS, Vc, Dc);
}
